// RQVAE_61838939128203
// MI455X (gfx1250) — matrix operand order repaired and checked by reading, not compiled, not run on hardware
//
#include <hip/hip_runtime.h>
#include <hip/hip_bf16.h>

typedef __attribute__((ext_vector_type(16))) _Float16 v16h;
typedef __attribute__((ext_vector_type(8)))  _Float16 v8h;
typedef __attribute__((ext_vector_type(8)))  float    v8f;
typedef __attribute__((ext_vector_type(4)))  unsigned int u32x4;
typedef __attribute__((ext_vector_type(8)))  int i32x8;
typedef __attribute__((ext_vector_type(4)))  int i32x4;

#if defined(__AMDGCN__) && __has_builtin(__builtin_amdgcn_tensor_load_to_lds) && \
    __has_builtin(__builtin_amdgcn_s_wait_tensorcnt)
#define USE_TDM 1
#else
#define USE_TDM 0
#endif

__device__ __forceinline__ v16h cat16(v8h lo, v8h hi) {
  return __builtin_shufflevector(lo, hi, 0,1,2,3,4,5,6,7,8,9,10,11,12,13,14,15);
}

// ---------------------------------------------------------------------------
// Weight prep: f32 conv weights -> f16 co-major GEMM layout Wt[co][k],
// k = ci*KH*KW + kh*KW + kw.
// HWIO=false: w is OIHW [COUT,CIN,KH,KW]; HWIO=true: w is [KH,KW,CIN,COUT]
// ---------------------------------------------------------------------------
template<int CIN, int COUT, int KH, int KW, bool HWIO>
__global__ void prep_weight(const float* __restrict__ w, _Float16* __restrict__ wt) {
  const int KTOT = CIN * KH * KW;
  const int n = COUT * KTOT;
  for (int e = blockIdx.x * blockDim.x + threadIdx.x; e < n; e += gridDim.x * blockDim.x) {
    int co = e / KTOT;  int k  = e - co * KTOT;
    int ci = k / (KH*KW); int rem = k - ci * (KH*KW);
    int kh = rem / KW;    int kw = rem - kh * KW;
    float v = HWIO ? w[(((size_t)kh*KW + kw)*CIN + ci)*COUT + co]
                   : w[(((size_t)co*CIN + ci)*KH + kh)*KW + kw];
    wt[e] = (_Float16)v;
  }
}

// Codebook prep: f16 copy (emb[l] is already [K,C] = co-major layout) + row norms
__global__ void emb_prep(const float* __restrict__ emb, _Float16* __restrict__ embh,
                         float* __restrict__ e2, int rows, int C) {
  int r = blockIdx.x * blockDim.x + threadIdx.x;
  if (r >= rows) return;
  float s = 0.0f;
  for (int c = 0; c < C; ++c) {
    float v = emb[(size_t)r * C + c];
    s += v * v;
    embh[(size_t)r * C + c] = (_Float16)v;
  }
  e2[r] = s;
}

// ---------------------------------------------------------------------------
// Generic implicit-GEMM conv on v_wmma_f32_16x16x32_f16.
// MODE 0: normal conv   ih = oh*STR - PAD + kh
// MODE 1: stride-2 transposed conv  ih = (oh + PAD - kh)/2, parity-masked.
// Block: 256 threads = 8 waves; tile = 128 pixels x 128 out-channels.
// Wave (wr=wave>>1, wc=wave&1) owns 32 px x 64 co: acc[2][4], 8 WMMA / K-step.
// B (weight) tile is DMA'd by the Tensor Data Mover into LDS with hardware
// row padding to the 144B conflict-free stride; A tile is an im2col VALU fill
// (does the f32->f16 conversion).  NPIX is always a multiple of 128.
// ---------------------------------------------------------------------------
template<int CIN, int COUT, int KH, int KW, int STR, int PAD,
         int HI, int WI, int HO, int WO, int MODE, bool RELU>
__global__ __launch_bounds__(256)
void conv_wmma(const float* __restrict__ in, const _Float16* __restrict__ wt,
               const float* __restrict__ bias, float* __restrict__ out, int nB) {
  constexpr int KTOT = CIN * KH * KW;
  constexpr int LS = 72;                  // LDS row stride in halves: 144B
  __shared__ __align__(16) _Float16 sA[128 * LS];
  __shared__ __align__(16) _Float16 sB[128 * LS];

  const int tid  = threadIdx.x;
  const int lane = tid & 31;
  const int wave = tid >> 5;
  const int wr   = wave >> 1;             // 0..3  (pixel dim, 32 px each)
  const int wc   = wave & 1;              // 0..1  (channel dim, 64 co each)
  const int p0   = blockIdx.x * 128;
  const int c0   = blockIdx.y * 128;
  const int NPIX = nB * HO * WO;
  (void)NPIX;

  v8f zero;
  #pragma unroll
  for (int t = 0; t < 8; ++t) zero[t] = 0.0f;
  v8f acc[2][4] = {{zero, zero, zero, zero}, {zero, zero, zero, zero}};

  // K-invariant per-thread A-fill decomposition: j = tid&31 is fixed.
  const int jA   = tid & 31;
  const int rowA = tid >> 5;              // + 8*i, i = 0..15

  for (int kk = 0; kk < KTOT; kk += 32) {
#if USE_TDM
    // ---- B tile via Tensor Data Mover: 128 rows x 64B, LDS-padded to 144B ----
    if (wave == 0) {
      unsigned lds_b = (unsigned)(uintptr_t)&sB[0];
      unsigned long long ga =
          (unsigned long long)(uintptr_t)wt + ((size_t)c0 * KTOT + (size_t)kk) * 2ull;
      int rem_rows  = COUT - c0; if (rem_rows > 128) rem_rows = 128;
      int rem_units = (KTOT - kk) >> 2;   // remaining 8B units in a row
      u32x4 g0 = {
        1u,                                    // count=1, user descriptor
        lds_b,                                 // lds_addr
        (unsigned)(ga & 0xffffffffull),        // global_addr[31:0]
        (unsigned)((ga >> 32) & 0x1ffffffull) | 0x80000000u  // [56:32] | type=2
      };
      i32x8 g1 = {
        (int)((3u << 16) | (1u << 20) | (3u << 22) | (19u << 25)),
        //    data_size=8B | pad_en    | every 16 DW | pad 20 DW (64B row -> 144B)
        (int)(((unsigned)rem_units & 0xffffu) << 16),          // tensor_dim0 lo
        (int)((((unsigned)rem_units >> 16) & 0xffffu) |
              (((unsigned)rem_rows & 0xffffu) << 16)),         // dim0 hi | dim1 lo
        (int)((((unsigned)rem_rows >> 16) & 0xffffu) |
              (8u << 16)),                                     // dim1 hi | tile_dim0=8
        128,                                                    // tile_dim1 = 128 rows
        (int)((unsigned)KTOT >> 2),                             // dim0 stride (8B units)
        0, 0
      };
      i32x4 gz4 = {0, 0, 0, 0};
      i32x8 gz8 = {0, 0, 0, 0, 0, 0, 0, 0};
      __builtin_amdgcn_tensor_load_to_lds(g0, g1, gz4, gz4, gz8, 0);
    }
#else
    {
      const int r0 = tid >> 5;            // B fill fallback: j fixed per thread
      int k = kk + jA;
      bool kv = (k < KTOT);
      #pragma unroll
      for (int i = 0; i < 16; ++i) {
        int r = r0 + i * 8;
        int co = c0 + r;
        _Float16 v = (_Float16)0.0f;
        if (kv && co < COUT) v = wt[(size_t)co * KTOT + k];
        sB[r * LS + jA] = v;
      }
    }
#endif
    // ---- A tile: cooperative im2col fill, 128 px x 32 K (f32 -> f16) ----
    {
      int k = kk + jA;
      int ci = 0, kh = 0, kw = 0;
      bool kv = (k < KTOT);
      if (kv) {
        ci = k / (KH*KW); int rem = k - ci * (KH*KW);
        kh = rem / KW;    kw  = rem - kh * KW;
      }
      #pragma unroll
      for (int i = 0; i < 16; ++i) {
        int row = rowA + i * 8;
        int p  = p0 + row;
        float v = 0.0f;
        if (kv) {
          int b  = p / (HO*WO); int pr = p - b * (HO*WO);
          int oh = pr / WO;     int ow = pr - oh * WO;
          int ih, iw; bool ok;
          if (MODE == 0) {
            ih = oh * STR - PAD + kh;
            iw = ow * STR - PAD + kw;
            ok = (ih >= 0) & (ih < HI) & (iw >= 0) & (iw < WI);
          } else {
            int th = oh + PAD - kh, tw = ow + PAD - kw;
            ok = (th >= 0) & (tw >= 0) & ((th & 1) == 0) & ((tw & 1) == 0);
            ih = th >> 1; iw = tw >> 1;
            ok = ok & (ih < HI) & (iw < WI);
          }
          if (ok) v = in[(((size_t)b * CIN + ci) * HI + ih) * WI + iw];
        }
        sA[row * LS + jA] = (_Float16)v;
      }
    }
#if USE_TDM
    if (wave == 0) __builtin_amdgcn_s_wait_tensorcnt(0);
#endif
    __syncthreads();

    // ---- per-wave fragment loads (aligned ds_read_b128 pairs) ----
    const int ar = wr * 32 + (lane & 15);
    const int ka = ((lane >> 4) & 1) * 16;  // A: 16 contiguous K per lane-half
    v16h afr[2];
    afr[0] = cat16(*(const v8h*)&sA[ ar       * LS + ka],
                   *(const v8h*)&sA[ ar       * LS + ka + 8]);
    afr[1] = cat16(*(const v8h*)&sA[(ar + 16) * LS + ka],
                   *(const v8h*)&sA[(ar + 16) * LS + ka + 8]);
    const int kb = ((lane >> 4) & 1) * 16;  // B: 16 contiguous K per lane-half
    v16h bfr[4];
    #pragma unroll
    for (int ni = 0; ni < 4; ++ni) {
      int br = wc * 64 + ni * 16 + (lane & 15);
      bfr[ni] = cat16(*(const v8h*)&sB[br * LS + kb],
                      *(const v8h*)&sB[br * LS + kb + 8]);
    }

    #pragma unroll
    for (int mi = 0; mi < 2; ++mi)
      #pragma unroll
      for (int ni = 0; ni < 4; ++ni)
        acc[mi][ni] = __builtin_amdgcn_wmma_f32_16x16x32_f16(
            false, afr[mi], false, bfr[ni], (short)0, acc[mi][ni], false, false);
    __syncthreads();
  }

  // ---- epilogue: bias (+ReLU), scatter to NCHW ----
  const int n  = lane & 15;
  const int mb = (lane >> 4) * 8;           // C/D layout: vgpr v -> M = mb + v
  #pragma unroll
  for (int mi = 0; mi < 2; ++mi) {
    #pragma unroll
    for (int ni = 0; ni < 4; ++ni) {
      int co = c0 + wc * 64 + ni * 16 + n;
      if (co >= COUT) continue;
      float bv = bias ? bias[co] : 0.0f;
      #pragma unroll
      for (int v = 0; v < 8; ++v) {
        int p = p0 + wr * 32 + mi * 16 + mb + v;
        int b  = p / (HO*WO); int pr = p - b * (HO*WO);
        float val = acc[mi][ni][v] + bv;
        if (RELU) val = fmaxf(val, 0.0f);
        out[((size_t)b * COUT + co) * (HO*WO) + pr] = val;
      }
    }
  }
}

// ---------------------------------------------------------------------------
// Residual quantization step: argmin_k (||E_k||^2 - 2 r.E_k), then
// residuals[l]=r; z_q += E[idx]; r -= E[idx]; codes[l]=idx.
// dots buffer is NCHW [B, K, H, W] (output of the 1x1 WMMA GEMM).
// ---------------------------------------------------------------------------
__global__ void argmin_update(const float* __restrict__ dots, const float* __restrict__ e2,
                              const float* __restrict__ embl, const float* __restrict__ rin,
                              float* __restrict__ rout, float* __restrict__ resid_out,
                              float* __restrict__ zq, float* __restrict__ codes,
                              int NP, int C, int HW, int Kc, int first) {
  int p = blockIdx.x * blockDim.x + threadIdx.x;
  if (p >= NP) return;
  int b = p / HW; int pr = p - b * HW;
  const float* dp = dots + (size_t)b * Kc * HW + pr;
  float best = 3.4e38f; int bi = 0;
  for (int k = 0; k < Kc; ++k) {
    float s = e2[k] - 2.0f * dp[(size_t)k * HW];
    if (s < best) { best = s; bi = k; }
  }
  codes[p] = (float)bi;
  const float* ev = embl + (size_t)bi * C;
  size_t base = (size_t)b * C * HW + pr;
  for (int c = 0; c < C; ++c) {
    size_t o = base + (size_t)c * HW;
    float rv = rin[o];
    resid_out[o] = rv;
    float e = ev[c];
    zq[o] = first ? e : (zq[o] + e);
    rout[o] = rv - e;
  }
}

// ---------------------------------------------------------------------------
extern "C" void kernel_launch(void* const* d_in, const int* in_sizes, int n_in,
                              void* d_out, int out_size, void* d_ws, size_t ws_size,
                              hipStream_t stream) {
  (void)in_sizes; (void)n_in; (void)out_size; (void)ws_size;
  const float* x      = (const float*)d_in[0];
  const float* enc_w0 = (const float*)d_in[1];
  const float* enc_b0 = (const float*)d_in[2];
  const float* enc_w1 = (const float*)d_in[3];
  const float* enc_b1 = (const float*)d_in[4];
  const float* enc_w2 = (const float*)d_in[5];
  const float* enc_b2 = (const float*)d_in[6];
  const float* proj_w = (const float*)d_in[7];
  const float* proj_b = (const float*)d_in[8];
  const float* dec_w0 = (const float*)d_in[9];
  const float* dec_b0 = (const float*)d_in[10];
  const float* dec_w1 = (const float*)d_in[11];
  const float* dec_b1 = (const float*)d_in[12];
  const float* dec_w2 = (const float*)d_in[13];
  const float* dec_b2 = (const float*)d_in[14];
  const float* out_w  = (const float*)d_in[15];
  const float* out_b  = (const float*)d_in[16];
  const float* emb    = (const float*)d_in[17];

  // d_out: recon | z | z_q | residuals[4] | codes[4]  (all f32)
  float* out   = (float*)d_out;
  float* recon = out;
  float* z     = out + 1572864;           // 8*3*256*256
  float* zq    = out + 3670016;           // + 8*256*32*32
  float* resid = out + 5767168;           // + 8*256*32*32
  float* codes = out + 14155776;          // + 4*8*256*32*32

  // Workspace carve-out
  char* wsb = (char*)d_ws;
  size_t off = 0;
  auto takeH = [&](size_t nelem) -> _Float16* {
    _Float16* p = (_Float16*)(wsb + off);
    off = (off + nelem * sizeof(_Float16) + 255) & ~(size_t)255;
    return p;
  };
  _Float16* wt_e0 = takeH((size_t)128 * 48);
  _Float16* wt_e1 = takeH((size_t)256 * 2048);
  _Float16* wt_e2 = takeH((size_t)512 * 4096);
  _Float16* wt_p  = takeH((size_t)256 * 512);
  _Float16* wt_d0 = takeH((size_t)512 * 4096);
  _Float16* wt_d1 = takeH((size_t)256 * 8192);
  _Float16* wt_d2 = takeH((size_t)128 * 4096);
  _Float16* wt_o  = takeH((size_t)3 * 1152);
  _Float16* embh  = takeH((size_t)4 * 1024 * 256);
  float* e2 = (float*)(wsb + off);
  off = (off + 4096 * sizeof(float) + 255) & ~(size_t)255;
  float* arena = (float*)(wsb + off);

  // Phase-based arena reuse (f32 activations)
  float* h1   = arena;                    // 8*128*128*128
  float* h2   = arena + 16777216;         // 8*256*64*64
  float* h3   = arena + 25165824;         // 8*512*32*32
  float* r    = arena;                    // quant phase: 8*256*32*32
  float* dots = arena + 2097152;          // quant phase: 8*1024*32*32
  float* d1   = arena;                    // 8*512*64*64
  float* d2   = arena + 16777216;         // 8*256*128*128
  float* d3   = arena + 50331648;         // 8*128*256*256

  // ---- weight / codebook prep ----
  prep_weight<3,   128, 4, 4, false><<<dim3(24),   256, 0, stream>>>(enc_w0, wt_e0);
  prep_weight<128, 256, 4, 4, false><<<dim3(512),  256, 0, stream>>>(enc_w1, wt_e1);
  prep_weight<256, 512, 4, 4, false><<<dim3(2048), 256, 0, stream>>>(enc_w2, wt_e2);
  prep_weight<512, 256, 1, 1, false><<<dim3(128),  256, 0, stream>>>(proj_w, wt_p);
  prep_weight<256, 512, 4, 4, true ><<<dim3(2048), 256, 0, stream>>>(dec_w0, wt_d0);
  prep_weight<512, 256, 4, 4, true ><<<dim3(2048), 256, 0, stream>>>(dec_w1, wt_d1);
  prep_weight<256, 128, 4, 4, true ><<<dim3(512),  256, 0, stream>>>(dec_w2, wt_d2);
  prep_weight<128, 3,   3, 3, false><<<dim3(8),    256, 0, stream>>>(out_w,  wt_o);
  emb_prep<<<dim3(16), 256, 0, stream>>>(emb, embh, e2, 4096, 256);

  // ---- encoder ----
  conv_wmma<3,   128, 4, 4, 2, 1, 256, 256, 128, 128, 0, true >
      <<<dim3(1024, 1), 256, 0, stream>>>(x,  wt_e0, enc_b0, h1, 8);
  conv_wmma<128, 256, 4, 4, 2, 1, 128, 128,  64,  64, 0, true >
      <<<dim3(256, 2),  256, 0, stream>>>(h1, wt_e1, enc_b1, h2, 8);
  conv_wmma<256, 512, 4, 4, 2, 1,  64,  64,  32,  32, 0, true >
      <<<dim3(64, 4),   256, 0, stream>>>(h2, wt_e2, enc_b2, h3, 8);
  conv_wmma<512, 256, 1, 1, 1, 0,  32,  32,  32,  32, 0, false>
      <<<dim3(64, 2),   256, 0, stream>>>(h3, wt_p, proj_b, z, 8);

  // ---- residual quantization: 4 x (WMMA distance GEMM + argmin/update) ----
  for (int l = 0; l < 4; ++l) {
    const float* rin = (l == 0) ? z : r;
    conv_wmma<256, 1024, 1, 1, 1, 0, 32, 32, 32, 32, 0, false>
        <<<dim3(64, 8), 256, 0, stream>>>(rin, embh + (size_t)l * 1024 * 256,
                                          nullptr, dots, 8);
    argmin_update<<<dim3(32), 256, 0, stream>>>(
        dots, e2 + (size_t)l * 1024, emb + (size_t)l * 1024 * 256,
        rin, r, resid + (size_t)l * 2097152, zq, codes + (size_t)l * 8192,
        8192, 256, 1024, 1024, (l == 0) ? 1 : 0);
  }

  // ---- decoder (straight-through forward value == z_q) ----
  conv_wmma<256, 512, 4, 4, 2, 1,  32,  32,  64,  64, 1, true >
      <<<dim3(256, 4),  256, 0, stream>>>(zq, wt_d0, dec_b0, d1, 8);
  conv_wmma<512, 256, 4, 4, 2, 1,  64,  64, 128, 128, 1, true >
      <<<dim3(1024, 2), 256, 0, stream>>>(d1, wt_d1, dec_b1, d2, 8);
  conv_wmma<256, 128, 4, 4, 2, 1, 128, 128, 256, 256, 1, true >
      <<<dim3(4096, 1), 256, 0, stream>>>(d2, wt_d2, dec_b2, d3, 8);
  conv_wmma<128, 3,   3, 3, 1, 1, 256, 256, 256, 256, 0, false>
      <<<dim3(4096, 1), 256, 0, stream>>>(d3, wt_o, out_b, recon, 8);
}
